// WindowAttention_73796128080367
// MI455X (gfx1250) — matrix operand order repaired and checked by reading, not compiled, not run on hardware
//
#include <hip/hip_runtime.h>

#define DIM 192
#define NH 6
#define HD 32
#define NN 64
#define SCALE 0.17677669529663687f  // 1/sqrt(32)

typedef __bf16 bf16;
typedef __attribute__((ext_vector_type(16))) bf16         v16bf;
typedef __attribute__((ext_vector_type(8)))  float        v8f;
typedef __attribute__((ext_vector_type(4)))  float        v4f;
typedef __attribute__((ext_vector_type(4)))  unsigned int v4u;
typedef __attribute__((ext_vector_type(2)))  unsigned int v2u;

union ABFrag {
  v16bf          bf;
  bf16           h[16];
  unsigned int   u32[8];
  unsigned short u16[16];
};

// native f32 -> bf16 (clang lowers to hw cvt on gfx1250)
__device__ __forceinline__ bf16 tobf(float f) { return (bf16)f; }

// pack two f32 -> dword of two bf16
__device__ __forceinline__ unsigned int pkbf(float lo, float hi) {
  unsigned int a = (unsigned int)__builtin_bit_cast(unsigned short, (bf16)lo);
  unsigned int b = (unsigned int)__builtin_bit_cast(unsigned short, (bf16)hi);
  return a | (b << 16);
}

// 16 bf16 from two 16B-aligned 8-element chunks (LDS or global)
__device__ __forceinline__ ABFrag load_frag_bf(const bf16* p, int off0, int off1) {
  ABFrag f;
  *(v4u*)&f.u32[0] = *(const v4u*)(p + off0);
  *(v4u*)&f.u32[4] = *(const v4u*)(p + off1);
  return f;
}

// weight-fragment loader, overloaded on weight storage type --------------------
__device__ __forceinline__ ABFrag load_wfrag(const bf16* p) {  // pre-converted ws
  return load_frag_bf(p, 0, 8);
}
__device__ __forceinline__ ABFrag load_wfrag(const float* __restrict__ p) {  // fallback
  v4f a = *(const v4f*)(p + 0);
  v4f b = *(const v4f*)(p + 4);
  v4f c = *(const v4f*)(p + 8);
  v4f d = *(const v4f*)(p + 12);
  float t[16] = {a.x, a.y, a.z, a.w, b.x, b.y, b.z, b.w,
                 c.x, c.y, c.z, c.w, d.x, d.y, d.z, d.w};
  ABFrag f;
#pragma unroll
  for (int i = 0; i < 16; ++i) f.h[i] = tobf(t[i]);
  return f;
}

__device__ __forceinline__ v8f wmma_bf16(const ABFrag& a, const ABFrag& b, v8f c) {
  return __builtin_amdgcn_wmma_f32_16x16x32_bf16(false, a.bf, false, b.bf,
                                                 (short)0, c, false, false);
}

// ---------------------------------------------------------------------------
// Weight pre-conversion: qkv_w (576x192) then proj_w (192x192) -> bf16 in ws
// ---------------------------------------------------------------------------
#define NQW (3 * DIM * DIM)  // 110592
#define NPW (DIM * DIM)      //  36864
__global__ __launch_bounds__(256) void cvt_weights(const float* __restrict__ qw,
                                                   const float* __restrict__ pw,
                                                   bf16* __restrict__ ws) {
  const int i = blockIdx.x * 256 + threadIdx.x;  // grid covers NQW+NPW exactly
  if (i < NQW) ws[i] = tobf(qw[i]);
  else         ws[i] = tobf(pw[i - NQW]);
}

// ---------------------------------------------------------------------------
// Kernel A: per (window b, head h): qkv (head slice) -> QK^T -> softmax(+mask)
//           -> P@V, pre-proj result written f32 to out.
// ---------------------------------------------------------------------------
template <typename WT>
__global__ __launch_bounds__(128) void wattn_qkv_sm_pv(
    const float* __restrict__ x, const float* __restrict__ mask,
    const WT* __restrict__ qkv_w, const float* __restrict__ qkv_b,
    float* __restrict__ out) {
  __shared__ bf16 s_x[NN * DIM];   // x tile bf16           24576 B
  __shared__ bf16 s_q[NN * HD];    // q*scale                4096 B
  __shared__ bf16 s_k[NN * HD];    // k                      4096 B
  __shared__ bf16 s_vT[HD * NN];   // v transposed [d][m]    4096 B
  __shared__ bf16 s_p[NN * NN];    // softmax probs          8192 B

  const int blk  = blockIdx.x;
  const int b    = blk / NH;
  const int h    = blk % NH;
  const int tid  = threadIdx.x;
  const int wave = tid >> 5;          // 4 waves == 4 M-tiles
  const int lane = tid & 31;
  const int l16  = lane & 15;
  const int hi8  = (lane >> 4) << 3;  // C-frag row offset: +0 / +8
  const int klo  = (lane >> 4) << 4;  // B-frag chunk select: +0 / +16
  const int row0 = wave * 16 + hi8;   // C-frag rows: row0 .. row0+7

  // ---- stage 0: x[b] -> bf16 LDS (packed dword stores) ---------------------
  const float* xb = x + (size_t)b * (NN * DIM);
  for (int i = tid; i < NN * DIM / 4; i += 128) {
    v4f v = *(const v4f*)(xb + 4 * i);
    v2u p = {pkbf(v.x, v.y), pkbf(v.z, v.w)};
    *(v2u*)(s_x + 4 * i) = p;
  }
  __syncthreads();

  // ---- stage 1: qkv GEMM head slice: 6 N-tiles (q0 q1 k0 k1 v0 v1) ---------
  {
    const int mrow = wave * 16 + l16;
    v8f acc[6] = {};
    for (int kk = 0; kk < 6; ++kk) {
      const int k0 = kk * 32;
      ABFrag a = load_frag_bf(s_x, mrow * DIM + k0 + klo, mrow * DIM + k0 + klo + 8);
#pragma unroll
      for (int nt = 0; nt < 6; ++nt) {
        const int wrow = (nt >> 1) * DIM + h * HD + (nt & 1) * 16 + l16;
        ABFrag bf = load_wfrag(qkv_w + (size_t)wrow * DIM + k0 + klo);
        acc[nt] = wmma_bf16(a, bf, acc[nt]);
      }
    }
#pragma unroll
    for (int nt = 0; nt < 6; ++nt) {
      const int wrow = (nt >> 1) * DIM + h * HD + (nt & 1) * 16 + l16;
      const float bias = qkv_b[wrow];
      const int c = (nt & 1) * 16 + l16;  // local channel 0..31
      if ((nt >> 1) == 0) {               // q: fold softmax scale
#pragma unroll
        for (int r = 0; r < 8; ++r)
          s_q[(row0 + r) * HD + c] = tobf((acc[nt][r] + bias) * SCALE);
      } else if ((nt >> 1) == 1) {        // k
#pragma unroll
        for (int r = 0; r < 8; ++r)
          s_k[(row0 + r) * HD + c] = tobf(acc[nt][r] + bias);
      } else {                            // v -> transposed, 8 contiguous m
        unsigned int pk[4];
#pragma unroll
        for (int r = 0; r < 4; ++r)
          pk[r] = pkbf(acc[nt][2 * r] + bias, acc[nt][2 * r + 1] + bias);
        *(v4u*)(s_vT + c * NN + row0) = *(v4u*)pk;
      }
    }
  }
  __syncthreads();

  // ---- stage 2: logits = (q*s) @ k^T + mask, softmax -> s_p ----------------
  {
    const int mrow = wave * 16 + l16;
    ABFrag a = load_frag_bf(s_q, mrow * HD + klo, mrow * HD + klo + 8);
    v8f acc[4] = {};
#pragma unroll
    for (int nt = 0; nt < 4; ++nt) {
      const int nrow = nt * 16 + l16;
      ABFrag bf = load_frag_bf(s_k, nrow * HD + klo, nrow * HD + klo + 8);
      acc[nt] = wmma_bf16(a, bf, acc[nt]);
    }
    const float* mrow_mask = mask + (size_t)(b & 63) * (NN * NN);  // nW == 64
    float vals[4][8];
#pragma unroll
    for (int nt = 0; nt < 4; ++nt)
#pragma unroll
      for (int r = 0; r < 8; ++r)
        vals[nt][r] = acc[nt][r] + mrow_mask[(row0 + r) * NN + nt * 16 + l16];

#pragma unroll
    for (int r = 0; r < 8; ++r) {
      float m = fmaxf(fmaxf(vals[0][r], vals[1][r]), fmaxf(vals[2][r], vals[3][r]));
      m = fmaxf(m, __shfl_xor(m, 1, 16));
      m = fmaxf(m, __shfl_xor(m, 2, 16));
      m = fmaxf(m, __shfl_xor(m, 4, 16));
      m = fmaxf(m, __shfl_xor(m, 8, 16));
      float s = 0.0f;
#pragma unroll
      for (int nt = 0; nt < 4; ++nt) {
        vals[nt][r] = __expf(vals[nt][r] - m);
        s += vals[nt][r];
      }
      s += __shfl_xor(s, 1, 16);
      s += __shfl_xor(s, 2, 16);
      s += __shfl_xor(s, 4, 16);
      s += __shfl_xor(s, 8, 16);
      const float inv = __builtin_amdgcn_rcpf(s);
#pragma unroll
      for (int nt = 0; nt < 4; ++nt)
        s_p[(row0 + r) * NN + nt * 16 + l16] = tobf(vals[nt][r] * inv);
    }
  }
  __syncthreads();

  // ---- stage 3: out_head = P @ V (pre-proj, f32 to global) -----------------
  {
    const int mrow = wave * 16 + l16;
    v8f acc[2] = {};
#pragma unroll
    for (int ks = 0; ks < 2; ++ks) {
      const int k0 = ks * 32;
      ABFrag a = load_frag_bf(s_p, mrow * NN + k0 + klo, mrow * NN + k0 + klo + 8);
#pragma unroll
      for (int nt = 0; nt < 2; ++nt) {
        const int d = nt * 16 + l16;
        ABFrag bf = load_frag_bf(s_vT, d * NN + k0 + klo, d * NN + k0 + klo + 8);
        acc[nt] = wmma_bf16(a, bf, acc[nt]);
      }
    }
    float* ob = out + (size_t)b * (NN * DIM) + h * HD;
#pragma unroll
    for (int nt = 0; nt < 2; ++nt) {
      const int d = nt * 16 + l16;
#pragma unroll
      for (int r = 0; r < 8; ++r) ob[(row0 + r) * DIM + d] = acc[nt][r];
    }
  }
}

// ---------------------------------------------------------------------------
// Kernel B: in-place output projection per window: io <- io @ proj_w^T + b
// ---------------------------------------------------------------------------
template <typename WT>
__global__ __launch_bounds__(256) void wattn_proj(
    float* __restrict__ io, const WT* __restrict__ proj_w,
    const float* __restrict__ proj_b) {
  __shared__ bf16 s_a[NN * DIM];  // 24576 B

  const int b    = blockIdx.x;
  const int tid  = threadIdx.x;
  const int wave = tid >> 5;
  const int lane = tid & 31;
  const int l16  = lane & 15;
  const int hi8  = (lane >> 4) << 3;
  const int klo  = (lane >> 4) << 4;

  float* xb = io + (size_t)b * (NN * DIM);
  for (int i = tid; i < NN * DIM / 4; i += 256) {
    v4f v = *(const v4f*)(xb + 4 * i);
    v2u p = {pkbf(v.x, v.y), pkbf(v.z, v.w)};
    *(v2u*)(s_a + 4 * i) = p;
  }
  __syncthreads();

  const int mt   = wave & 3;
  const int ng   = (wave >> 2) * 6;  // waves 0-3: nt 0..5, waves 4-7: 6..11
  const int mrow = mt * 16 + l16;
  const int row0 = mt * 16 + hi8;

  v8f acc[6] = {};
  for (int kk = 0; kk < 6; ++kk) {
    const int k0 = kk * 32;
    ABFrag a = load_frag_bf(s_a, mrow * DIM + k0 + klo, mrow * DIM + k0 + klo + 8);
#pragma unroll
    for (int j = 0; j < 6; ++j) {
      const int n0 = (ng + j) * 16 + l16;
      ABFrag bf = load_wfrag(proj_w + (size_t)n0 * DIM + k0 + klo);
      acc[j] = wmma_bf16(a, bf, acc[j]);
    }
  }
#pragma unroll
  for (int j = 0; j < 6; ++j) {
    const int c = (ng + j) * 16 + l16;
    const float bias = proj_b[c];
#pragma unroll
    for (int r = 0; r < 8; ++r) xb[(row0 + r) * DIM + c] = acc[j][r] + bias;
  }
}

extern "C" void kernel_launch(void* const* d_in, const int* in_sizes, int n_in,
                              void* d_out, int out_size, void* d_ws, size_t ws_size,
                              hipStream_t stream) {
  const float* x      = (const float*)d_in[0];
  const float* mask   = (const float*)d_in[1];
  const float* qkv_w  = (const float*)d_in[2];
  const float* qkv_b  = (const float*)d_in[3];
  const float* proj_w = (const float*)d_in[4];
  const float* proj_b = (const float*)d_in[5];
  float* out = (float*)d_out;

  const int Bw = in_sizes[0] / (NN * DIM);  // 4096 windows

  if (ws_size >= (size_t)(NQW + NPW) * sizeof(bf16)) {
    // fast path: one-time weight conversion into scratch, hot loops load bf16
    bf16* wsq = (bf16*)d_ws;
    bf16* wsp = wsq + NQW;
    cvt_weights<<<(NQW + NPW) / 256, 256, 0, stream>>>(qkv_w, proj_w, wsq);
    wattn_qkv_sm_pv<bf16><<<Bw * NH, 128, 0, stream>>>(x, mask, wsq, qkv_b, out);
    wattn_proj<bf16><<<Bw, 256, 0, stream>>>(out, wsp, proj_b);
  } else {
    // fallback: convert weight fragments inline from f32
    wattn_qkv_sm_pv<float><<<Bw * NH, 128, 0, stream>>>(x, mask, qkv_w, qkv_b, out);
    wattn_proj<float><<<Bw, 256, 0, stream>>>(out, proj_w, proj_b);
  }
}
